// QuantLinearLayer_2224793060101
// MI455X (gfx1250) — matrix operand order repaired and checked by reading, not compiled, not run on hardware
//
#include <hip/hip_runtime.h>

// ---- types -----------------------------------------------------------------
typedef __attribute__((ext_vector_type(16))) __bf16 v16bf;
typedef __attribute__((ext_vector_type(8)))  __bf16 v8bf;
typedef __attribute__((ext_vector_type(4)))  __bf16 v4bf;
typedef __attribute__((ext_vector_type(8)))  float  v8f;

// ---- problem/tile geometry -------------------------------------------------
constexpr int MDIM = 4096, NDIM = 4096, KDIM = 4096;
constexpr int BM = 128, BN = 128, BK = 32;
constexpr int LDA = 40;   // bf16 units: 80B row pitch (16B aligned), padded vs 32
constexpr int LDB = 40;   // transposed W tile pitch

__device__ __forceinline__ float rq(float x) {
    // bits_quant: round-to-nearest-even to multiples of 2^-8
    return rintf(x * 256.0f) * 0.00390625f;
}

__global__ __launch_bounds__(256)
void qlinear_bf16x3_wmma(const float* __restrict__ A,
                         const float* __restrict__ W,
                         const float* __restrict__ bias,
                         float* __restrict__ out)
{
    __shared__ __bf16 sAhi[BM * LDA];
    __shared__ __bf16 sAlo[BM * LDA];
    __shared__ __bf16 sBhi[BN * LDB];   // transposed: [n][k]
    __shared__ __bf16 sBlo[BN * LDB];

    const int tid   = threadIdx.x;
    const int lane  = tid & 31;
    const int wid   = tid >> 5;          // 0..7
    const int half  = lane >> 4;         // 0 or 1
    const int lm    = lane & 15;
    const int waveM = wid >> 1;          // 0..3  (rows: waveM*32)
    const int waveN = wid & 1;           // 0..1  (cols: waveN*64)

    const int rowBase = blockIdx.y * BM;
    const int colBase = blockIdx.x * BN;

    // per-thread staging assignments (fixed across K-steps)
    int aRow[4], aC4[4], wK[4], wN[4];
    #pragma unroll
    for (int i = 0; i < 4; ++i) {
        int idx = i * 256 + tid;          // 1024 float4s per tile
        aRow[i] = idx >> 3;               // A: 8 float4 per 32-col row
        aC4[i]  = idx & 7;
        wK[i]   = idx >> 5;               // W: 32 float4 per 128-col row
        wN[i]   = (idx & 31) * 4;
    }

    v8f acc[2][4];
    #pragma unroll
    for (int mi = 0; mi < 2; ++mi)
        #pragma unroll
        for (int ni = 0; ni < 4; ++ni)
            acc[mi][ni] = (v8f){0.f,0.f,0.f,0.f,0.f,0.f,0.f,0.f};

    // ---- prologue: prefetch K-step 0 into registers ----
    float4 aReg[4], wReg[4];
    #pragma unroll
    for (int i = 0; i < 4; ++i) {
        aReg[i] = *(const float4*)(A + (size_t)(rowBase + aRow[i]) * KDIM + aC4[i] * 4);
        wReg[i] = *(const float4*)(W + (size_t)(wK[i]) * NDIM + colBase + wN[i]);
    }

    for (int kb = 0; kb < KDIM; kb += BK) {
        // ---- convert current registers -> bf16 hi/lo in LDS ----
        #pragma unroll
        for (int i = 0; i < 4; ++i) {
            const float* f = (const float*)&aReg[i];
            v4bf h, l;
            #pragma unroll
            for (int j = 0; j < 4; ++j) {
                __bf16 hb = (__bf16)f[j];
                h[j] = hb;
                l[j] = (__bf16)(f[j] - (float)hb);
            }
            int o = aRow[i] * LDA + aC4[i] * 4;       // 8B aligned
            *(v4bf*)(&sAhi[o]) = h;
            *(v4bf*)(&sAlo[o]) = l;
        }
        #pragma unroll
        for (int i = 0; i < 4; ++i) {
            const float* f = (const float*)&wReg[i];
            #pragma unroll
            for (int j = 0; j < 4; ++j) {
                __bf16 hb = (__bf16)f[j];
                int o = (wN[i] + j) * LDB + wK[i];    // transposed scatter
                sBhi[o] = hb;
                sBlo[o] = (__bf16)(f[j] - (float)hb);
            }
        }

        // ---- prefetch next K-step's fp32 tiles (in flight during WMMAs) ----
        if (kb + BK < KDIM) {
            int kn = kb + BK;
            #pragma unroll
            for (int i = 0; i < 4; ++i) {
                aReg[i] = *(const float4*)(A + (size_t)(rowBase + aRow[i]) * KDIM + kn + aC4[i] * 4);
                wReg[i] = *(const float4*)(W + (size_t)(kn + wK[i]) * NDIM + colBase + wN[i]);
            }
        }

        __syncthreads();

        // ---- A fragments (ISA 7.12.2 16-bit A 16x32 layout):
        // lane<16 : elems 0..7 = K[0..7],  elems 8..15 = K[16..23]
        // lane>=16: elems 0..7 = K[8..15], elems 8..15 = K[24..31]
        v16bf ahi[2], alo[2];
        #pragma unroll
        for (int mi = 0; mi < 2; ++mi) {
            int r  = waveM * 32 + mi * 16 + lm;
            int k0 = half * 8;
            int k1 = 16 + half * 8;
            v8bf h0 = *(const v8bf*)(&sAhi[r * LDA + k0]);
            v8bf h1 = *(const v8bf*)(&sAhi[r * LDA + k1]);
            v8bf l0 = *(const v8bf*)(&sAlo[r * LDA + k0]);
            v8bf l1 = *(const v8bf*)(&sAlo[r * LDA + k1]);
            #pragma unroll
            for (int e = 0; e < 8; ++e) {
                ahi[mi][e] = h0[e]; ahi[mi][e + 8] = h1[e];
                alo[mi][e] = l0[e]; alo[mi][e + 8] = l1[e];
            }
        }

        // ---- B fragments (32x16, 16-bit): lane<16 -> N=lm,K=0..7,16..23;
        //      lane>=16 -> N=lm,K=8..15,24..31  (matches A layout)
        #pragma unroll
        for (int ni = 0; ni < 4; ++ni) {
            int c  = waveN * 64 + ni * 16 + lm;
            int kk = half * 8;
            v8bf bh0 = *(const v8bf*)(&sBhi[c * LDB + kk]);
            v8bf bh1 = *(const v8bf*)(&sBhi[c * LDB + kk + 16]);
            v8bf bl0 = *(const v8bf*)(&sBlo[c * LDB + kk]);
            v8bf bl1 = *(const v8bf*)(&sBlo[c * LDB + kk + 16]);
            v16bf bhi, blo;
            #pragma unroll
            for (int e = 0; e < 8; ++e) {
                bhi[e] = bh0[e]; bhi[e + 8] = bh1[e];
                blo[e] = bl0[e]; blo[e + 8] = bl1[e];
            }
            #pragma unroll
            for (int mi = 0; mi < 2; ++mi) {
                // fp32-emulating bf16x3: hi*hi + hi*lo + lo*hi, f32 accumulate
                acc[mi][ni] = __builtin_amdgcn_wmma_f32_16x16x32_bf16(
                    false, ahi[mi], false, bhi, (short)0, acc[mi][ni], false, false);
                acc[mi][ni] = __builtin_amdgcn_wmma_f32_16x16x32_bf16(
                    false, ahi[mi], false, blo, (short)0, acc[mi][ni], false, false);
                acc[mi][ni] = __builtin_amdgcn_wmma_f32_16x16x32_bf16(
                    false, alo[mi], false, bhi, (short)0, acc[mi][ni], false, false);
            }
        }

        __syncthreads();   // LDS consumed; safe to overwrite next iteration
    }

    // ---- epilogue: out = rq(acc) + rq(bias)  (final rq is identity) ----
    // C/D layout: lane<16 -> N=lm, VGPR r = M=r; lane>=16 -> N=lm, M=8+r
    #pragma unroll
    for (int ni = 0; ni < 4; ++ni) {
        int col  = colBase + waveN * 64 + ni * 16 + lm;
        float bq = rq(bias[col]);
        #pragma unroll
        for (int mi = 0; mi < 2; ++mi) {
            int row0 = rowBase + waveM * 32 + mi * 16 + half * 8;
            #pragma unroll
            for (int r = 0; r < 8; ++r) {
                float v = rq(acc[mi][ni][r]) + bq;
                out[(size_t)(row0 + r) * NDIM + col] = v;
            }
        }
    }
}

extern "C" void kernel_launch(void* const* d_in, const int* in_sizes, int n_in,
                              void* d_out, int out_size, void* d_ws, size_t ws_size,
                              hipStream_t stream) {
    const float* A    = (const float*)d_in[0];   // inputs [4096,4096]
    const float* W    = (const float*)d_in[1];   // w      [4096,4096]
    const float* bias = (const float*)d_in[2];   // b      [4096]
    float* out        = (float*)d_out;

    dim3 grid(NDIM / BN, MDIM / BM);   // (32, 32)
    dim3 block(256);
    qlinear_bf16x3_wmma<<<grid, block, 0, stream>>>(A, W, bias, out);
}
